// NNCLR_13486197310091
// MI455X (gfx1250) — matrix operand order repaired and checked by reading, not compiled, not run on hardware
//
#include <hip/hip_runtime.h>

// ---------------------------------------------------------------------------
// NNCLR on MI455X (gfx1250): bf16 WMMA (v_wmma_f32_16x16x32_bf16), LDS-staged
// B tiles via async global->LDS copies (ASYNCcnt, double buffered), fused
// argmax / online-logsumexp reductions on C fragments, wave32.
// ---------------------------------------------------------------------------

typedef __attribute__((ext_vector_type(16))) __bf16 v16bf;
typedef __attribute__((ext_vector_type(8)))  __bf16 v8bf;
typedef __attribute__((ext_vector_type(8)))  float  v8f;

static constexpr int kB = 4096;    // batch
static constexpr int kD = 256;     // feature dim
static constexpr int kQ = 32768;   // queue length
static constexpr float kInvT = 10.0f;  // 1 / TEMPERATURE

static constexpr int kSuper   = 64;            // columns staged per LDS buffer
static constexpr int kTileB   = kSuper * kD * 2;  // 32 KB per buffer
static constexpr int kNNChunk = kQ / 4;        // 8192 cols per NN chunk
static constexpr int kLsChunk = kB / 4;        // 1024 cols per loss chunk

#define DEVINL static __device__ __forceinline__

DEVINL unsigned short f2bf(float f) {
  unsigned int u = __float_as_uint(f);
  u += 0x7FFFu + ((u >> 16) & 1u);        // round-to-nearest-even
  return (unsigned short)(u >> 16);
}
DEVINL unsigned int pack2(float a, float b) {
  return (unsigned int)f2bf(a) | ((unsigned int)f2bf(b) << 16);
}

// A fragment, 16-bit A 16x32 (ISA 7.12.2): lane L holds row M=L&15.
DEVINL v16bf load_a_frag(const unsigned short* rowp, int k0, int half) {
  v8bf lo = *(const v8bf*)(rowp + k0 + half * 8);
  v8bf hi = *(const v8bf*)(rowp + k0 + 16 + half * 8);
  return __builtin_shufflevector(lo, hi, 0, 1, 2, 3, 4, 5, 6, 7,
                                 8, 9, 10, 11, 12, 13, 14, 15);
}

// B fragment from LDS: two runs of 8 bf16 at sp and sp+16 (two ds_load_b128).
DEVINL v16bf lds_b_frag(const unsigned short* sp) {
  v8bf lo = *(const v8bf*)sp;
  v8bf hi = *(const v8bf*)(sp + 16);
  return __builtin_shufflevector(lo, hi, 0, 1, 2, 3, 4, 5, 6, 7,
                                 8, 9, 10, 11, 12, 13, 14, 15);
}

// gfx1250 async global -> LDS copy, 16B per lane, tracked by ASYNCcnt.
DEVINL void async_b128(unsigned lds_off, unsigned goff, const void* base) {
  asm volatile("global_load_async_to_lds_b128 %0, %1, %2"
               :
               : "v"(lds_off), "v"(goff), "s"(base)
               : "memory");
}
DEVINL void wait_async8()  { asm volatile("s_wait_asynccnt 8" ::: "memory"); }
DEVINL void wait_async0()  { asm volatile("s_wait_asynccnt 0" ::: "memory"); }

// ---------------------------------------------------------------------------
// Kernel 1: L2-normalize p1/p2 rows.  One wave per row (8 floats per lane).
// ---------------------------------------------------------------------------
__global__ void nnclr_normalize(const float* __restrict__ p1,
                                const float* __restrict__ p2,
                                unsigned short* __restrict__ bf1,
                                unsigned short* __restrict__ bf2,
                                float* __restrict__ newq) {
  const int lane = threadIdx.x & 31;
  const int wave = threadIdx.x >> 5;
  const int row  = blockIdx.x * (blockDim.x >> 5) + wave;
  const float* src = blockIdx.y ? p2 : p1;
  unsigned short* dst = blockIdx.y ? bf2 : bf1;

  const float4* s = (const float4*)(src + (size_t)row * kD) + lane * 2;
  float4 a = s[0], b = s[1];
  float ss = a.x * a.x + a.y * a.y + a.z * a.z + a.w * a.w +
             b.x * b.x + b.y * b.y + b.z * b.z + b.w * b.w;
#pragma unroll
  for (int m = 16; m >= 1; m >>= 1) ss += __shfl_xor(ss, m, 32);
  const float inv = rsqrtf(ss);
  a.x *= inv; a.y *= inv; a.z *= inv; a.w *= inv;
  b.x *= inv; b.y *= inv; b.z *= inv; b.w *= inv;

  uint4 u;
  u.x = pack2(a.x, a.y); u.y = pack2(a.z, a.w);
  u.z = pack2(b.x, b.y); u.w = pack2(b.z, b.w);
  ((uint4*)(dst + (size_t)row * kD))[lane] = u;

  if (blockIdx.y == 0) {
    float4* o = (float4*)(newq + (size_t)row * kD) + lane * 2;
    o[0] = a; o[1] = b;
  }
}

// ---------------------------------------------------------------------------
// Kernel 2: queue -> bf16 copy + new_queue[B:Q) = queue[0:Q-B).
// ---------------------------------------------------------------------------
__global__ void nnclr_queue_prep(const float* __restrict__ q,
                                 unsigned short* __restrict__ qbf,
                                 float* __restrict__ newq) {
  const int lane = threadIdx.x & 31;
  const int wave = threadIdx.x >> 5;
  const int row  = blockIdx.x * (blockDim.x >> 5) + wave;

  const float4* s = (const float4*)(q + (size_t)row * kD) + lane * 2;
  float4 a = s[0], b = s[1];
  uint4 u;
  u.x = pack2(a.x, a.y); u.y = pack2(a.z, a.w);
  u.z = pack2(b.x, b.y); u.w = pack2(b.z, b.w);
  ((uint4*)(qbf + (size_t)row * kD))[lane] = u;

  if (row < kQ - kB) {
    float4* o = (float4*)(newq + (size_t)(row + kB) * kD) + lane * 2;
    o[0] = a; o[1] = b;
  }
}

// ---------------------------------------------------------------------------
// Kernel 3: NN search.  8 waves/WG, 2 row-tiles/wave (256 rows/WG).  B super-
// tiles (64 cols = 32 KB, contiguous) double-buffered in LDS via async copies
// shared by all waves.  grid = (B/256, 4 chunks, 2 sides), block = 256.
// ---------------------------------------------------------------------------
__global__ void __launch_bounds__(256)
nnclr_nn_partial(const unsigned short* __restrict__ bf1,
                 const unsigned short* __restrict__ bf2,
                 const unsigned short* __restrict__ qbf,
                 float* __restrict__ pval, int* __restrict__ pidx) {
  __shared__ unsigned short smem[2 * kSuper * kD];   // 64 KB
  const int lane = threadIdx.x & 31;
  const int wave = threadIdx.x >> 5;
  const int half = lane >> 4;
  const int nlo  = lane & 15;
  const int chunk = blockIdx.y;
  const int side  = blockIdx.z;
  const int rowbase = blockIdx.x * 256 + wave * 32;   // two 16-row tiles

  const unsigned short* A = side ? bf2 : bf1;
  const unsigned short* ar0 = A + (size_t)(rowbase + nlo) * kD;
  const unsigned short* ar1 = ar0 + 16 * kD;
  v16bf af0[8], af1[8];
#pragma unroll
  for (int t = 0; t < 8; ++t) {
    af0[t] = load_a_frag(ar0, t * 32, half);
    af1[t] = load_a_frag(ar1, t * 32, half);
  }

  float bm0[8], bm1[8]; int bi0[8], bi1[8];
#pragma unroll
  for (int i = 0; i < 8; ++i) {
    bm0[i] = -INFINITY; bm1[i] = -INFINITY; bi0[i] = 0; bi1[i] = 0;
  }

  const unsigned lds_base = (unsigned)(uintptr_t)(void*)smem;
  const unsigned c0b = (unsigned)(chunk * kNNChunk) * (unsigned)(kD * 2);
  const int nst = kNNChunk / kSuper;                  // 128

  auto issue = [&](int st, int buf) {
    const unsigned gb = c0b + (unsigned)st * (unsigned)kTileB;
    const unsigned lb = lds_base + (unsigned)buf * (unsigned)kTileB;
#pragma unroll
    for (int k = 0; k < 8; ++k) {                     // wave's 4 KB share
      unsigned idx = ((unsigned)(wave * 8 + k) * 32u + (unsigned)lane) * 16u;
      async_b128(lb + idx, gb + idx, qbf);
    }
  };

  issue(0, 0);
  for (int st = 0; st < nst; ++st) {
    const int buf = st & 1;
    if (st + 1 < nst) { issue(st + 1, buf ^ 1); wait_async8(); }
    else              { wait_async0(); }
    __syncthreads();                                  // tile ready for all waves
#pragma unroll
    for (int ct = 0; ct < 4; ++ct) {
      const unsigned short* bp =
          &smem[((size_t)buf * kSuper + ct * 16 + nlo) * kD + half * 8];
      v8f a0 = {}, a1 = {};
#pragma unroll
      for (int t = 0; t < 8; ++t) {
        v16bf bf = lds_b_frag(bp + t * 32);
        a0 = __builtin_amdgcn_wmma_f32_16x16x32_bf16(
            false, af0[t], false, bf, (short)0, a0, false, false);
        a1 = __builtin_amdgcn_wmma_f32_16x16x32_bf16(
            false, af1[t], false, bf, (short)0, a1, false, false);
      }
      const int col = chunk * kNNChunk + st * kSuper + ct * 16 + nlo;
#pragma unroll
      for (int i = 0; i < 8; ++i) {
        float v0 = a0[i], v1 = a1[i];
        if (v0 > bm0[i]) { bm0[i] = v0; bi0[i] = col; }
        if (v1 > bm1[i]) { bm1[i] = v1; bi1[i] = col; }
      }
    }
    __syncthreads();                                  // done reading before reuse
  }

#pragma unroll
  for (int i = 0; i < 8; ++i) {
#pragma unroll
    for (int m = 1; m <= 8; m <<= 1) {
      float ov0 = __shfl_xor(bm0[i], m, 32); int oi0 = __shfl_xor(bi0[i], m, 32);
      float ov1 = __shfl_xor(bm1[i], m, 32); int oi1 = __shfl_xor(bi1[i], m, 32);
      if (ov0 > bm0[i] || (ov0 == bm0[i] && oi0 < bi0[i])) { bm0[i] = ov0; bi0[i] = oi0; }
      if (ov1 > bm1[i] || (ov1 == bm1[i] && oi1 < bi1[i])) { bm1[i] = ov1; bi1[i] = oi1; }
    }
    if (nlo == 0) {
      int r0 = rowbase + i + half * 8;
      size_t o0 = ((size_t)side * 4 + chunk) * kB + r0;
      pval[o0] = bm0[i]; pidx[o0] = bi0[i];
      pval[o0 + 16] = bm1[i]; pidx[o0 + 16] = bi1[i];   // second tile: rows +16
    }
  }
}

// ---------------------------------------------------------------------------
// Kernel 4: merge 4 chunk-partials per row and gather winning bf16 queue row.
// ---------------------------------------------------------------------------
__global__ void nnclr_nn_gather(const float* __restrict__ pval,
                                const int* __restrict__ pidx,
                                const unsigned short* __restrict__ qbf,
                                unsigned short* __restrict__ nn1,
                                unsigned short* __restrict__ nn2) {
  const int lane = threadIdx.x & 31;
  const int wave = threadIdx.x >> 5;
  const int row  = blockIdx.x * (blockDim.x >> 5) + wave;
  const int side = blockIdx.y;

  int best = 0;
  if (lane == 0) {
    float bv = -INFINITY; int bix = 0;
#pragma unroll
    for (int c = 0; c < 4; ++c) {
      size_t o = ((size_t)side * 4 + c) * kB + row;
      float v = pval[o]; int ix = pidx[o];
      if (v > bv || (v == bv && ix < bix)) { bv = v; bix = ix; }
    }
    best = bix;
  }
  best = __shfl(best, 0, 32);
  unsigned short* dst = side ? nn2 : nn1;
  ((uint4*)(dst + (size_t)row * kD))[lane] =
      ((const uint4*)(qbf + (size_t)best * kD))[lane];
}

// ---------------------------------------------------------------------------
// Kernel 5: four [B,B] logit blocks, LDS-staged B, fused online logsumexp +
// diag capture, column-chunked.  grid = (B/256, 4 chunks, 4 passes).
// ---------------------------------------------------------------------------
__global__ void __launch_bounds__(256)
nnclr_loss_partial(const unsigned short* __restrict__ nn1,
                   const unsigned short* __restrict__ p2bf,
                   const unsigned short* __restrict__ nn2,
                   const unsigned short* __restrict__ p1bf,
                   float* __restrict__ pm, float* __restrict__ ps,
                   float* __restrict__ pd) {
  __shared__ unsigned short smem[2 * kSuper * kD];   // 64 KB
  const int lane = threadIdx.x & 31;
  const int wave = threadIdx.x >> 5;
  const int half = lane >> 4;
  const int nlo  = lane & 15;
  const int chunk = blockIdx.y;
  const int pass  = blockIdx.z;   // 0: nn1*p2^T 1: p2*nn1^T 2: nn2*p1^T 3: p1*nn2^T
  const int rowbase = blockIdx.x * 256 + wave * 32;

  const unsigned short* A;
  const unsigned short* Bm;
  if      (pass == 0) { A = nn1;  Bm = p2bf; }
  else if (pass == 1) { A = p2bf; Bm = nn1;  }
  else if (pass == 2) { A = nn2;  Bm = p1bf; }
  else                { A = p1bf; Bm = nn2;  }

  const unsigned short* ar0 = A + (size_t)(rowbase + nlo) * kD;
  const unsigned short* ar1 = ar0 + 16 * kD;
  v16bf af0[8], af1[8];
#pragma unroll
  for (int t = 0; t < 8; ++t) {
    af0[t] = load_a_frag(ar0, t * 32, half);
    af1[t] = load_a_frag(ar1, t * 32, half);
  }

  float m0[8], s0[8], d0[8], m1[8], s1[8], d1[8];
#pragma unroll
  for (int i = 0; i < 8; ++i) {
    m0[i] = -INFINITY; s0[i] = 0.f; d0[i] = 0.f;
    m1[i] = -INFINITY; s1[i] = 0.f; d1[i] = 0.f;
  }

  const unsigned lds_base = (unsigned)(uintptr_t)(void*)smem;
  const unsigned c0b = (unsigned)(chunk * kLsChunk) * (unsigned)(kD * 2);
  const int nst = kLsChunk / kSuper;                 // 16

  auto issue = [&](int st, int buf) {
    const unsigned gb = c0b + (unsigned)st * (unsigned)kTileB;
    const unsigned lb = lds_base + (unsigned)buf * (unsigned)kTileB;
#pragma unroll
    for (int k = 0; k < 8; ++k) {
      unsigned idx = ((unsigned)(wave * 8 + k) * 32u + (unsigned)lane) * 16u;
      async_b128(lb + idx, gb + idx, Bm);
    }
  };

  issue(0, 0);
  for (int st = 0; st < nst; ++st) {
    const int buf = st & 1;
    if (st + 1 < nst) { issue(st + 1, buf ^ 1); wait_async8(); }
    else              { wait_async0(); }
    __syncthreads();
#pragma unroll
    for (int ct = 0; ct < 4; ++ct) {
      const unsigned short* bp =
          &smem[((size_t)buf * kSuper + ct * 16 + nlo) * kD + half * 8];
      v8f a0 = {}, a1 = {};
#pragma unroll
      for (int t = 0; t < 8; ++t) {
        v16bf bf = lds_b_frag(bp + t * 32);
        a0 = __builtin_amdgcn_wmma_f32_16x16x32_bf16(
            false, af0[t], false, bf, (short)0, a0, false, false);
        a1 = __builtin_amdgcn_wmma_f32_16x16x32_bf16(
            false, af1[t], false, bf, (short)0, a1, false, false);
      }
      const int col = chunk * kLsChunk + st * kSuper + ct * 16 + nlo;
#pragma unroll
      for (int i = 0; i < 8; ++i) {
        float v = a0[i] * kInvT;
        int grow = rowbase + i + half * 8;
        if (col == grow) d0[i] = v;
        if (v > m0[i]) { s0[i] = s0[i] * __expf(m0[i] - v) + 1.0f; m0[i] = v; }
        else           { s0[i] += __expf(v - m0[i]); }
        float w = a1[i] * kInvT;
        if (col == grow + 16) d1[i] = w;
        if (w > m1[i]) { s1[i] = s1[i] * __expf(m1[i] - w) + 1.0f; m1[i] = w; }
        else           { s1[i] += __expf(w - m1[i]); }
      }
    }
    __syncthreads();
  }

#pragma unroll
  for (int i = 0; i < 8; ++i) {
#pragma unroll
    for (int msk = 1; msk <= 8; msk <<= 1) {
      float om = __shfl_xor(m0[i], msk, 32), os = __shfl_xor(s0[i], msk, 32);
      float od = __shfl_xor(d0[i], msk, 32);
      float nm = fmaxf(m0[i], om);
      s0[i] = s0[i] * __expf(m0[i] - nm) + os * __expf(om - nm);
      m0[i] = nm; d0[i] += od;
      om = __shfl_xor(m1[i], msk, 32); os = __shfl_xor(s1[i], msk, 32);
      od = __shfl_xor(d1[i], msk, 32);
      nm = fmaxf(m1[i], om);
      s1[i] = s1[i] * __expf(m1[i] - nm) + os * __expf(om - nm);
      m1[i] = nm; d1[i] += od;
    }
    if (nlo == 0) {
      int r0 = rowbase + i + half * 8;
      size_t o0 = ((size_t)pass * 4 + chunk) * kB + r0;
      pm[o0] = m0[i]; ps[o0] = s0[i]; pd[o0] = d0[i];
      pm[o0 + 16] = m1[i]; ps[o0 + 16] = s1[i]; pd[o0 + 16] = d1[i];
    }
  }
}

// ---------------------------------------------------------------------------
// Kernel 6: merge loss chunk-partials -> final loss[4B].
// ---------------------------------------------------------------------------
__global__ void nnclr_loss_merge(const float* __restrict__ pm,
                                 const float* __restrict__ ps,
                                 const float* __restrict__ pd,
                                 float* __restrict__ loss) {
  const int idx = blockIdx.x * blockDim.x + threadIdx.x;   // pass*B + row
  const int pass = idx >> 12;
  const int row  = idx & (kB - 1);
  float m = -INFINITY, s = 0.f, dg = 0.f;
#pragma unroll
  for (int c = 0; c < 4; ++c) {
    size_t o = ((size_t)pass * 4 + c) * kB + row;
    float om = pm[o], os = ps[o];
    float nm = fmaxf(m, om);
    s = s * __expf(m - nm) + os * __expf(om - nm);
    m = nm;
    dg += pd[o];
  }
  loss[idx] = m + logf(s) - dg;
}

// ---------------------------------------------------------------------------
extern "C" void kernel_launch(void* const* d_in, const int* in_sizes, int n_in,
                              void* d_out, int out_size, void* d_ws, size_t ws_size,
                              hipStream_t stream) {
  (void)in_sizes; (void)n_in; (void)out_size; (void)ws_size;
  const float* p1 = (const float*)d_in[0];
  const float* p2 = (const float*)d_in[1];
  const float* fq = (const float*)d_in[2];

  float* loss = (float*)d_out;            // [4B]
  float* newq = loss + (size_t)4 * kB;    // [Q, D]

  char* w = (char*)d_ws;
  size_t off = 0;
  auto take = [&](size_t bytes) -> void* {
    void* p = w + off;
    off = (off + bytes + 255) & ~(size_t)255;
    return p;
  };
  unsigned short* bf1 = (unsigned short*)take((size_t)kB * kD * 2);  // 2 MB
  unsigned short* bf2 = (unsigned short*)take((size_t)kB * kD * 2);  // 2 MB
  unsigned short* qbf = (unsigned short*)take((size_t)kQ * kD * 2);  // 16 MB
  unsigned short* nn1 = (unsigned short*)take((size_t)kB * kD * 2);  // 2 MB
  unsigned short* nn2 = (unsigned short*)take((size_t)kB * kD * 2);  // 2 MB
  float* pval = (float*)take((size_t)2 * 4 * kB * 4);                // 128 KB
  int*   pidx = (int*)take((size_t)2 * 4 * kB * 4);                  // 128 KB
  float* pm   = (float*)take((size_t)4 * 4 * kB * 4);                // 256 KB
  float* ps   = (float*)take((size_t)4 * 4 * kB * 4);                // 256 KB
  float* pd   = (float*)take((size_t)4 * 4 * kB * 4);                // 256 KB

  nnclr_normalize<<<dim3(kB / 8, 2), 256, 0, stream>>>(p1, p2, bf1, bf2, newq);
  nnclr_queue_prep<<<dim3(kQ / 8), 256, 0, stream>>>(fq, qbf, newq);
  nnclr_nn_partial<<<dim3(kB / 256, 4, 2), 256, 0, stream>>>(bf1, bf2, qbf, pval, pidx);
  nnclr_nn_gather<<<dim3(kB / 8, 2), 256, 0, stream>>>(pval, pidx, qbf, nn1, nn2);
  nnclr_loss_partial<<<dim3(kB / 256, 4, 4), 256, 0, stream>>>(nn1, bf2, nn2, bf1,
                                                               pm, ps, pd);
  nnclr_loss_merge<<<dim3(4 * kB / 256), 256, 0, stream>>>(pm, ps, pd, loss);
}
